// CrossAttention_88029649699603
// MI455X (gfx1250) — matrix operand order repaired and checked by reading, not compiled, not run on hardware
//
#include <hip/hip_runtime.h>
#include <hip/hip_bf16.h>

// ---- problem constants (from reference) ----
constexpr int kB = 4;
constexpr int kC = 256;
constexpr int kN = 4096;                 // 64*64 tokens
constexpr float kScaleLog2e = 0.125f * 1.44269504088896f; // (1/sqrt(64)) * log2(e)

typedef __attribute__((ext_vector_type(16))) _Float16 v16h;
typedef __attribute__((ext_vector_type(8)))  _Float16 v8h;
typedef __attribute__((ext_vector_type(8)))  float    v8f;

union Frag32 { v16h v; v8h h[2]; };      // 32-byte A/B fragment (two 16B chunks)
union HalfU  { v8h h; int i[4]; _Float16 e[8]; };

__device__ __forceinline__ v8f wmma16(v16h a, v16h b, v8f c) {
  // D = A(16x32 f16) * B(32x16 f16) + C(16x16 f32)
  return __builtin_amdgcn_wmma_f32_16x16x32_f16(false, a, false, b, (short)0, c, false, false);
}

// CDNA5 async global->LDS DMA (ASYNCcnt-tracked), per ISA 15.18.3 / CDNA5 ISA
__device__ __forceinline__ void async_b128(unsigned lds_addr, const _Float16* g) {
  asm volatile("global_load_async_to_lds_b128 %0, %1, off"
               :: "v"(lds_addr), "v"(g) : "memory");
}
__device__ __forceinline__ void wait_async0() {
  asm volatile("s_wait_asynccnt 0x0" ::: "memory");
}

// LDS tile rows are 32 halves (4 chunks of 8 halves); chunk c of row r is stored
// at physical column (c ^ ((r>>2)&3)) -> conflict-free 16-lane fragment reads.
// A-fragment (16x32, K contiguous per row): lane needs chunks {hi, hi+2}.
__device__ __forceinline__ v16h lds_read_a(const _Float16* base, int row, int hi) {
  int sw = (row >> 2) & 3;
  Frag32 f;
  f.h[0] = *(const v8h*)(base + (row << 5) + (((hi)     ^ sw) << 3));
  f.h[1] = *(const v8h*)(base + (row << 5) + (((hi + 2) ^ sw) << 3));
  return f.v;
}
// B-fragment (32x16, K contiguous per column): lane needs chunks {hi, hi+2}.
__device__ __forceinline__ v16h lds_read_b(const _Float16* base, int row, int hi) {
  int sw = (row >> 2) & 3;
  Frag32 f;
  f.h[0] = *(const v8h*)(base + (row << 5) + (((hi)     ^ sw) << 3));
  f.h[1] = *(const v8h*)(base + (row << 5) + (((hi + 2) ^ sw) << 3));
  return f.v;
}

// ---------------------------------------------------------------------------
// Kernel 1: convert the four CxC f32 weight matrices to f16 (layout kept [d][c])
// ---------------------------------------------------------------------------
__global__ void k_cvt_weights(const float* __restrict__ s0, const float* __restrict__ s1,
                              const float* __restrict__ s2, const float* __restrict__ s3,
                              _Float16* __restrict__ dst) {
  int i = blockIdx.x * blockDim.x + threadIdx.x;   // 4 * 65536 total
  int m = i >> 16, off = i & 65535;
  const float* s = (m == 0) ? s0 : (m == 1) ? s1 : (m == 2) ? s2 : s3;
  dst[i] = (_Float16)s[off];
}

// ---------------------------------------------------------------------------
// Kernel 2: transpose feat [b][c][n] f32 -> [b][n][c] f16 via LDS 64x64 tiles
// ---------------------------------------------------------------------------
__global__ void k_transpose_cvt(const float* __restrict__ fa, const float* __restrict__ fb,
                                _Float16* __restrict__ ta, _Float16* __restrict__ tb) {
  __shared__ float tile[64][65];
  int bid = blockIdx.x;                       // 2 * kB * (kC/64) * (kN/64)
  int tilesPerImg = (kC / 64) * (kN / 64);    // 256
  int img  = bid / tilesPerImg;               // 0..7
  int tidx = bid % tilesPerImg;
  int c0 = (tidx / (kN / 64)) * 64;
  int n0 = (tidx % (kN / 64)) * 64;
  int b  = img & 3;
  const float* src = (img < 4) ? fa : fb;
  _Float16*    dst = (img < 4) ? ta : tb;
  src += (size_t)b * kC * kN;
  dst += (size_t)b * kN * kC;

  int tx = threadIdx.x & 63, ty = threadIdx.x >> 6; // 64 x 4
#pragma unroll
  for (int i = 0; i < 16; ++i) {
    int c = i * 4 + ty;
    tile[c][tx] = src[(size_t)(c0 + c) * kN + n0 + tx];
  }
  __syncthreads();
#pragma unroll
  for (int i = 0; i < 16; ++i) {
    int n = i * 4 + ty;
    dst[(size_t)(n0 + n) * kC + c0 + tx] = (_Float16)tile[tx][n];
  }
}

// ---------------------------------------------------------------------------
// Kernel 3: QKV projections, async-LDS double-buffered tiles.
//   Q[n][d] = sum_c a[n][c] Wq[d][c] + bq[d]    (stored [b][n][d] f16)
//   K[n][d] likewise from feat_B                 (stored [b][n][d] f16)
//   Vt[d][n] = sum_c b[n][c] Wv[d][c] + bv[d]    (stored [b][d][n] f16)
// Per 32-c step: DMA W tile [256][32] (16KB, shared by all 8 waves) and token
// tile [64][32] (4KB) into LDS; WMMA fragments come from ds_load_b128.
// grid: kB * (kN/64) * 3 ; block 256 (8 waves). Wave -> 16 tokens x 8 d-tiles.
// ---------------------------------------------------------------------------
__global__ void __launch_bounds__(256) k_qkv(
    const _Float16* __restrict__ featAt, const _Float16* __restrict__ featBt,
    const _Float16* __restrict__ Wq16, const _Float16* __restrict__ Wk16,
    const _Float16* __restrict__ Wv16,
    const float* __restrict__ bq, const float* __restrict__ bk,
    const float* __restrict__ bv,
    _Float16* __restrict__ Qd, _Float16* __restrict__ Kd,
    _Float16* __restrict__ Vtd) {
  __shared__ _Float16 sW[2][256][32];   // 2 x 16 KB
  __shared__ _Float16 sF[2][64][32];    // 2 x  4 KB  (total 40 KB)

  int bid  = blockIdx.x;
  int proj = bid % 3;
  int rest = bid / 3;
  int nt   = rest % (kN / 64);
  int b    = rest / (kN / 64);

  int t = threadIdx.x;
  int lane = t & 31, wave = t >> 5;
  int lm = lane & 15;
  int hi = (lane >= 16) ? 1 : 0;
  int nloc = (wave & 3) * 16;            // wave's token rows within the tile
  int nb   = nt * 64 + nloc;             // global token base for this wave
  int dgrp = wave >> 2;                  // 0/1 -> d 0..127 / 128..255

  const _Float16* feat = ((proj == 0) ? featAt : featBt) + (size_t)b * kN * kC;
  const _Float16* Wm   = (proj == 0) ? Wq16 : (proj == 1) ? Wk16 : Wv16;

  unsigned wL[2] = { (unsigned)(size_t)&sW[0][0][0], (unsigned)(size_t)&sW[1][0][0] };
  unsigned fL[2] = { (unsigned)(size_t)&sF[0][0][0], (unsigned)(size_t)&sF[1][0][0] };

  auto issue = [&](int buf, int c0) {
#pragma unroll
    for (int j = 0; j < 4; ++j) {        // W tile: 1024 chunks of 16B, 4/thread
      int i = t + 256 * j;
      int d = i >> 2, c = i & 3;
      async_b128(wL[buf] + (unsigned)(((d << 5) + ((c ^ ((d >> 2) & 3)) << 3)) << 1),
                 Wm + (size_t)d * kC + c0 + c * 8);
    }
    {                                    // token tile: 256 chunks of 16B, 1/thread
      int n = t >> 2, c = t & 3;
      async_b128(fL[buf] + (unsigned)(((n << 5) + ((c ^ ((n >> 2) & 3)) << 3)) << 1),
                 feat + (size_t)(nt * 64 + n) * kC + c0 + c * 8);
    }
  };

  v8f acc[8];
#pragma unroll
  for (int j = 0; j < 8; ++j) acc[j] = (v8f)0.0f;

  issue(0, 0);
  for (int s = 0; s < kC / 32; ++s) {
    int cur = s & 1;
    wait_async0();
    __syncthreads();
    if (s + 1 < kC / 32) issue(cur ^ 1, (s + 1) * 32);

    const _Float16* wb = &sW[cur][0][0];
    const _Float16* fb = &sF[cur][0][0];
    if (proj < 2) {
      // D[M=n][N=d]: A = token rows, B = W^T columns
      v16h af = lds_read_a(fb, nloc + lm, hi);
#pragma unroll
      for (int j = 0; j < 8; ++j)
        acc[j] = wmma16(af, lds_read_b(wb, dgrp * 128 + j * 16 + lm, hi), acc[j]);
    } else {
      // D[M=d][N=n]: A = Wv rows, B = token columns
      v16h bf2 = lds_read_b(fb, nloc + lm, hi);
#pragma unroll
      for (int j = 0; j < 8; ++j)
        acc[j] = wmma16(lds_read_a(wb, dgrp * 128 + j * 16 + lm, hi), bf2, acc[j]);
    }
  }

  if (proj < 2) {
    const float* bias = (proj == 0) ? bq : bk;
    _Float16*    dst  = (proj == 0) ? Qd : Kd;
#pragma unroll
    for (int j = 0; j < 8; ++j) {
      int d = dgrp * 128 + j * 16 + lm;
      float bv_ = bias[d];
#pragma unroll
      for (int r = 0; r < 8; ++r) {
        int n = nb + r + hi * 8;
        dst[((size_t)b * kN + n) * kC + d] = (_Float16)(acc[j][r] + bv_);
      }
    }
  } else {
#pragma unroll
    for (int j = 0; j < 8; ++j) {
#pragma unroll
      for (int r = 0; r < 8; ++r) {
        int d = dgrp * 128 + j * 16 + r + hi * 8;
        Vtd[((size_t)b * kC + d) * kN + nb + lm] = (_Float16)(acc[j][r] + bv[d]);
      }
    }
  }
}

// ---------------------------------------------------------------------------
// Kernel 4: flash attention with async-LDS double-buffered K/V staging.
//   Block = 128 threads (4 waves) = 64 q-tokens; wave owns 16 q-tokens.
//   Scores computed transposed S^T[k][q]; online softmax per column via
//   shfl_xor(16); P converted in-register to a 32x16 B-fragment for P*V.
// ---------------------------------------------------------------------------
__global__ void __launch_bounds__(128) k_attn(const _Float16* __restrict__ Qd,
                                              const _Float16* __restrict__ Kd,
                                              const _Float16* __restrict__ Vtd,
                                              _Float16* __restrict__ Od) {
  __shared__ _Float16 sK[2][32][256];   // 2 x 16 KB
  __shared__ _Float16 sV[2][256][32];   // 2 x 16 KB   (total 64 KB)

  int t = threadIdx.x;
  int bid = blockIdx.x;
  int nt  = bid % (kN / 64);
  int b   = bid / (kN / 64);
  int lane = t & 31, wave = t >> 5;
  int lm = lane & 15;
  int hi = (lane >= 16) ? 1 : 0;
  bool hiL = hi != 0;
  int qb = nt * 64 + wave * 16;

  const _Float16* Qb = Qd  + (size_t)b * kN * kC;
  const _Float16* Kb = Kd  + (size_t)b * kN * kC;
  const _Float16* Vb = Vtd + (size_t)b * kC * kN;

  unsigned kL[2] = { (unsigned)(size_t)&sK[0][0][0], (unsigned)(size_t)&sK[1][0][0] };
  unsigned vL[2] = { (unsigned)(size_t)&sV[0][0][0], (unsigned)(size_t)&sV[1][0][0] };

  auto issue = [&](int buf, int k0) {
    const _Float16* Kg = Kb + (size_t)k0 * kC;    // 32 contiguous rows of 256
    const _Float16* Vg = Vb + k0;                 // 256 rows of 32, stride kN
#pragma unroll
    for (int j = 0; j < 8; ++j) {                 // K tile: 1024 chunks of 16B
      int i = t + 128 * j;
      int r = i >> 4, c = i & 15;
      async_b128(kL[buf] + (unsigned)(((r << 8) + ((c ^ (r & 15)) << 3)) << 1),
                 Kg + (size_t)r * kC + c * 8);
    }
#pragma unroll
    for (int j = 0; j < 8; ++j) {                 // V tile: 1024 chunks of 16B
      int i = t + 128 * j;
      int d = i >> 2, c = i & 3;
      async_b128(vL[buf] + (unsigned)(((d << 5) + ((c ^ ((d >> 2) & 3)) << 3)) << 1),
                 Vg + (size_t)d * kN + c * 8);
    }
  };

  // cache Q as B-fragments: B[d][q] = Q[q][d], one 32B load each
  v16h qf[8];
#pragma unroll
  for (int ds = 0; ds < 8; ++ds)
    qf[ds] = *(const v16h*)(Qb + (size_t)(qb + lm) * kC + ds * 32 + hi * 16);

  v8f o[16];
#pragma unroll
  for (int dt = 0; dt < 16; ++dt) o[dt] = (v8f)0.0f;
  float m = -1e30f, l = 0.0f;

  issue(0, 0);                                    // prologue prefetch

  for (int it = 0; it < kN / 32; ++it) {
    int cur = it & 1;
    wait_async0();          // my DMA for buf[cur] (and older) done
    __syncthreads();        // everyone's DMA landed; prev consumers finished
    if (it + 1 < kN / 32) issue(cur ^ 1, (it + 1) * 32);

    const _Float16* kbase = &sK[cur][0][0];
    const _Float16* vbase = &sV[cur][0][0];

    // ---- scores: S^T tiles (rows = k, cols = q) ----
    v8f s0 = (v8f)0.0f, s1 = (v8f)0.0f;
#pragma unroll
    for (int ds = 0; ds < 8; ++ds) {
      int ck = ds * 4 + 2 * hi;                   // logical 8-half chunk index
      Frag32 ka;
      ka.h[0] = *(const v8h*)(kbase + (lm << 8) + ((ck ^ lm) << 3));
      ka.h[1] = *(const v8h*)(kbase + (lm << 8) + (((ck + 1) ^ lm) << 3));
      s0 = wmma16(ka.v, qf[ds], s0);
      ka.h[0] = *(const v8h*)(kbase + ((16 + lm) << 8) + ((ck ^ lm) << 3));
      ka.h[1] = *(const v8h*)(kbase + ((16 + lm) << 8) + (((ck + 1) ^ lm) << 3));
      s1 = wmma16(ka.v, qf[ds], s1);
    }

    // ---- online softmax over the 32 new k for each q-column ----
    float mt = -1e30f;
#pragma unroll
    for (int r = 0; r < 8; ++r) mt = fmaxf(mt, fmaxf(s0[r], s1[r]));
    mt = fmaxf(mt, __shfl_xor(mt, 16, 32));
    float mnew = fmaxf(m, mt);
    float corr = __builtin_amdgcn_exp2f(kScaleLog2e * (m - mnew));
    l *= corr;
#pragma unroll
    for (int dt = 0; dt < 16; ++dt) o[dt] *= corr;

    HalfU p0, p1;
    float rs = 0.0f;
#pragma unroll
    for (int r = 0; r < 8; ++r) {
      float e0 = __builtin_amdgcn_exp2f(kScaleLog2e * (s0[r] - mnew));
      float e1 = __builtin_amdgcn_exp2f(kScaleLog2e * (s1[r] - mnew));
      rs += e0 + e1;
      p0.e[r] = (_Float16)e0;
      p1.e[r] = (_Float16)e1;
    }
    rs += __shfl_xor(rs, 16, 32);
    l += rs;
    m = mnew;

    // ---- build P^T B-fragment (32k x 16q): swap halves across lane groups ----
    HalfU p0x, p1x;
#pragma unroll
    for (int w = 0; w < 4; ++w) {
      p0x.i[w] = __shfl_xor(p0.i[w], 16, 32);
      p1x.i[w] = __shfl_xor(p1.i[w], 16, 32);
    }
    Frag32 pb;
    pb.h[0] = hiL ? p1x.h : p0.h;   // k rows 0..7  of this lane group
    pb.h[1] = hiL ? p1.h  : p0x.h;  // k rows 8..15 of this lane group

    // ---- O^T[d][q] += V^T(d x 32k) * P^T(32k x q) ----
#pragma unroll
    for (int dt = 0; dt < 16; ++dt) {
      int d = dt * 16 + lm;
      int sw = (d >> 2) & 3;
      Frag32 va;
      va.h[0] = *(const v8h*)(vbase + (d << 5) + (((2 * hi) ^ sw) << 3));
      va.h[1] = *(const v8h*)(vbase + (d << 5) + (((2 * hi + 1) ^ sw) << 3));
      o[dt] = wmma16(va.v, pb.v, o[dt]);
    }
  }

  float invl = 1.0f / l;
#pragma unroll
  for (int dt = 0; dt < 16; ++dt) {
    v8h oh;
#pragma unroll
    for (int r = 0; r < 8; ++r) oh[r] = (_Float16)(o[dt][r] * invl);
    // lane holds 8 consecutive d for token q -> single 16B store, O stored [n][d]
    *(v8h*)(Od + ((size_t)b * kN + qb + lm) * kC + dt * 16 + hi * 8) = oh;
  }
}

// ---------------------------------------------------------------------------
// Kernel 5: output projection into [B][C][N] f32 (= [B,C,H,W]), async-LDS tiles.
//   out^T[co][n] = sum_d Wo[co][d] * O[n][d] + bo[co]
// grid: kB * (kN/64) ; block 256 (8 waves).
// ---------------------------------------------------------------------------
__global__ void __launch_bounds__(256) k_oproj(const _Float16* __restrict__ Od,
                                               const _Float16* __restrict__ Wo16,
                                               const float* __restrict__ bo,
                                               float* __restrict__ out) {
  __shared__ _Float16 sW[2][256][32];   // Wo rows
  __shared__ _Float16 sF[2][64][32];    // O token rows

  int bid = blockIdx.x;
  int nt  = bid % (kN / 64);
  int b   = bid / (kN / 64);
  int t = threadIdx.x;
  int lane = t & 31, wave = t >> 5;
  int lm = lane & 15;
  int hi = (lane >= 16) ? 1 : 0;
  int nloc = (wave & 3) * 16;
  int nb   = nt * 64 + nloc;
  int cgrp = wave >> 2;

  const _Float16* Ob = Od + (size_t)b * kN * kC;

  unsigned wL[2] = { (unsigned)(size_t)&sW[0][0][0], (unsigned)(size_t)&sW[1][0][0] };
  unsigned fL[2] = { (unsigned)(size_t)&sF[0][0][0], (unsigned)(size_t)&sF[1][0][0] };

  auto issue = [&](int buf, int d0) {
#pragma unroll
    for (int j = 0; j < 4; ++j) {
      int i = t + 256 * j;
      int d = i >> 2, c = i & 3;
      async_b128(wL[buf] + (unsigned)(((d << 5) + ((c ^ ((d >> 2) & 3)) << 3)) << 1),
                 Wo16 + (size_t)d * kC + d0 + c * 8);
    }
    {
      int n = t >> 2, c = t & 3;
      async_b128(fL[buf] + (unsigned)(((n << 5) + ((c ^ ((n >> 2) & 3)) << 3)) << 1),
                 Ob + (size_t)(nt * 64 + n) * kC + d0 + c * 8);
    }
  };

  v8f acc[8];
#pragma unroll
  for (int j = 0; j < 8; ++j) acc[j] = (v8f)0.0f;

  issue(0, 0);
  for (int s = 0; s < kC / 32; ++s) {
    int cur = s & 1;
    wait_async0();
    __syncthreads();
    if (s + 1 < kC / 32) issue(cur ^ 1, (s + 1) * 32);

    const _Float16* wb = &sW[cur][0][0];
    const _Float16* fb = &sF[cur][0][0];
    // D[M=co][N=n]: A = Wo rows, B = O token columns
    v16h bf2 = lds_read_b(fb, nloc + lm, hi);
#pragma unroll
    for (int j = 0; j < 8; ++j)
      acc[j] = wmma16(lds_read_a(wb, cgrp * 128 + j * 16 + lm, hi), bf2, acc[j]);
  }

#pragma unroll
  for (int j = 0; j < 8; ++j) {
#pragma unroll
    for (int r = 0; r < 8; ++r) {
      int co = cgrp * 128 + j * 16 + r + hi * 8;
      out[((size_t)b * kC + co) * kN + nb + lm] = acc[j][r] + bo[co];
    }
  }
}

// ---------------------------------------------------------------------------
extern "C" void kernel_launch(void* const* d_in, const int* in_sizes, int n_in,
                              void* d_out, int out_size, void* d_ws, size_t ws_size,
                              hipStream_t stream) {
  const float* fA = (const float*)d_in[0];
  const float* fB = (const float*)d_in[1];
  const float* Wq = (const float*)d_in[2];
  const float* bq = (const float*)d_in[3];
  const float* Wk = (const float*)d_in[4];
  const float* bk = (const float*)d_in[5];
  const float* Wv = (const float*)d_in[6];
  const float* bv = (const float*)d_in[7];
  const float* Wo = (const float*)d_in[8];
  const float* bo = (const float*)d_in[9];
  float* out = (float*)d_out;

  const size_t BNC = (size_t)kB * kN * kC;   // 4,194,304 elements
  _Float16* featAt = (_Float16*)d_ws;
  _Float16* featBt = featAt + BNC;
  _Float16* W16    = featBt + BNC;           // 4 * C*C halves
  _Float16* Wq16   = W16;
  _Float16* Wk16   = W16 + 1 * kC * kC;
  _Float16* Wv16   = W16 + 2 * kC * kC;
  _Float16* Wo16   = W16 + 3 * kC * kC;
  _Float16* Qh     = W16 + 4 * kC * kC;
  _Float16* Kh     = Qh + BNC;
  _Float16* Vth    = Kh + BNC;
  _Float16* Oh     = featAt;                 // reuse: featAt dead after k_qkv

  // 1) weights f32 -> f16
  k_cvt_weights<<<(4 * kC * kC) / 256, 256, 0, stream>>>(Wq, Wk, Wv, Wo, W16);
  // 2) feats [b][c][n] f32 -> [b][n][c] f16 (LDS transpose)
  k_transpose_cvt<<<2 * kB * (kC / 64) * (kN / 64), 256, 0, stream>>>(fA, fB, featAt, featBt);
  // 3) Q, K, V^T projections (WMMA + async-LDS tiles)
  k_qkv<<<kB * (kN / 64) * 3, 256, 0, stream>>>(featAt, featBt, Wq16, Wk16, Wv16,
                                                bq, bk, bv, Qh, Kh, Vth);
  // 4) flash attention (WMMA + async-LDS pipeline + online softmax)
  k_attn<<<kB * (kN / 64), 128, 0, stream>>>(Qh, Kh, Vth, Oh);
  // 5) output projection -> [B,C,H,W] f32 (WMMA + async-LDS tiles)
  k_oproj<<<kB * (kN / 64), 256, 0, stream>>>(Oh, Wo16, bo, out);
}
